// MultiHeadAttention_2740189135558
// MI455X (gfx1250) — matrix operand order repaired and checked by reading, not compiled, not run on hardware
//
#include <hip/hip_runtime.h>

// ---------------------------------------------------------------------------
// Multi-head attention for MI455X (gfx1250), bf16 WMMA + f32 accumulate.
// cvt(f32->bf16) -> Q/K/V projections (pipelined WMMA GEMM) ->
// flash-attention (LDS-staged K/V via async global->LDS, WMMA +
// online softmax) -> output projection (WMMA GEMM, f32 out).
// Intermediates (~50 MB bf16) live in d_ws and fit in the 192 MB L2.
// ---------------------------------------------------------------------------

#define DEV __device__ __forceinline__

typedef __attribute__((ext_vector_type(16))) __bf16 v16bf;
typedef __attribute__((ext_vector_type(8)))  float  v8f;
typedef int v4i_ __attribute__((vector_size(16)));   // int4 for async builtin

constexpr int Bsz = 2;        // batch
constexpr int Sq  = 2048;     // sequence
constexpr int Dm  = 1024;     // d_model
constexpr int Hn  = 16;       // heads
constexpr int Dh  = 64;       // head dim
constexpr int BS  = Bsz * Sq; // 4096 token rows

// ---- CDNA5 async global->LDS path (probe-verified on this toolchain) ------
#if defined(__has_builtin)
#if __has_builtin(__builtin_amdgcn_global_load_async_to_lds_b128) && \
    __has_builtin(__builtin_amdgcn_s_wait_asynccnt)
#define KV_ASYNC 1
#endif
#endif

DEV void copy16_to_lds(const unsigned short* g, unsigned short* l) {
#ifdef KV_ASYNC
  __builtin_amdgcn_global_load_async_to_lds_b128(
      (__attribute__((address_space(1))) v4i_*)g,
      (__attribute__((address_space(3))) v4i_*)l,
      /*offset=*/0, /*cpol=*/0);
#else
  *(uint4*)l = *(const uint4*)g;
#endif
}

DEV void wait_stage_done() {
#ifdef KV_ASYNC
  __builtin_amdgcn_s_wait_asynccnt(0);
#endif
}

// ---- bf16 helpers (raw-bit storage as ushort) -----------------------------

DEV unsigned short f32_to_bf16(float f) {
  unsigned int u = __float_as_uint(f);
  u += 0x7FFFu + ((u >> 16) & 1u);       // round-to-nearest-even
  return (unsigned short)(u >> 16);
}

// ---- WMMA fragment loaders (contiguous k order, same for A and B) -----------------
// A-frag 16x32 bf16: lane<16 -> row=lane,   K = k0+0..15
//                    lane>=16-> row=lane-16,K = k0+16..31
DEV v16bf load_frag_a(const unsigned short* p, int ld, int row0, int k0) {
  const int lane = threadIdx.x & 31;
  const int half = lane >> 4;
  const __bf16* r =
      (const __bf16*)(p + (size_t)(row0 + (lane & 15)) * ld + k0 + half * 16);
  v16bf out;
#pragma unroll
  for (int e = 0; e < 8; ++e) out[e] = r[e];
#pragma unroll
  for (int e = 0; e < 8; ++e) out[8 + e] = r[8 + e];
  return out;
}

// B-frag 32x16 bf16 with B[k][n] = src[n0+n][k0+k]  (src row-major [N][K]).
// lane<16 -> col=lane, K=k0+0..15 ; lane>=16 -> col=lane-16, K=k0+16..31.
DEV v16bf load_frag_b(const unsigned short* p, int ld, int n0, int k0) {
  const int lane = threadIdx.x & 31;
  const __bf16* r =
      (const __bf16*)(p + (size_t)(n0 + (lane & 15)) * ld + k0 + (lane >> 4) * 16);
  v16bf out;
#pragma unroll
  for (int e = 0; e < 16; ++e) out[e] = r[e];
  return out;
}

DEV v8f wmma_bf16(v16bf a, v16bf b, v8f c) {
  return __builtin_amdgcn_wmma_f32_16x16x32_bf16(
      /*neg_a=*/false, a, /*neg_b=*/false, b,
      /*c_mod=*/(short)0, c, /*reuse_a=*/false, /*reuse_b=*/false);
}

// ---- fp32 -> bf16 conversion, 4 elements/thread ---------------------------

__global__ __launch_bounds__(256)
void cvt_f32_bf16_kernel(const float4* __restrict__ in,
                         unsigned long long* __restrict__ out, int n4) {
  int i = blockIdx.x * blockDim.x + threadIdx.x;
  if (i < n4) {
    float4 v = in[i];
    unsigned long long r =
        (unsigned long long)f32_to_bf16(v.x) |
        ((unsigned long long)f32_to_bf16(v.y) << 16) |
        ((unsigned long long)f32_to_bf16(v.z) << 32) |
        ((unsigned long long)f32_to_bf16(v.w) << 48);
    out[i] = r;
  }
}

// ---- GEMM: Y[M=BS][N=Dm] = A[M][K=Dm] @ W[N][K]^T -------------------------
// MODE 0: Q  bf16 [B,H,S,Dh], pre-scaled by 1/sqrt(Dh)
// MODE 1: K  bf16 [B,H,S,Dh]
// MODE 2: Vt bf16 [B,H,Dh,S]   (transposed so P@V B-frags are contiguous)
// MODE 3: f32 row-major [M][N]  (final output projection)
// Each wave computes a 16x64 strip; K-loop is software-pipelined (register
// double-buffer) so loads of iteration k+1 overlap the WMMAs of iteration k.
template <int MODE>
__global__ __launch_bounds__(128)
void gemm_bf16_kernel(const unsigned short* __restrict__ A,
                      const unsigned short* __restrict__ W,
                      unsigned short* __restrict__ outb,
                      float* __restrict__ outf) {
  const int wave = threadIdx.x >> 5;
  const int m0 = blockIdx.x * 16;
  const int n0 = blockIdx.y * 256 + wave * 64;

  v8f zero = {};
  v8f acc[4];
#pragma unroll
  for (int j = 0; j < 4; ++j) acc[j] = zero;

  v16bf a_cur = load_frag_a(A, Dm, m0, 0);
  v16bf b_cur[4];
#pragma unroll
  for (int j = 0; j < 4; ++j) b_cur[j] = load_frag_b(W, Dm, n0 + 16 * j, 0);

  for (int k0 = 0; k0 < Dm; k0 += 32) {
    v16bf a_nxt = a_cur;
    v16bf b_nxt[4];
#pragma unroll
    for (int j = 0; j < 4; ++j) b_nxt[j] = b_cur[j];
    const int kn = k0 + 32;
    if (kn < Dm) {
      __builtin_prefetch(A + (size_t)m0 * Dm + kn + 32, 0, 0);
      a_nxt = load_frag_a(A, Dm, m0, kn);
#pragma unroll
      for (int j = 0; j < 4; ++j) b_nxt[j] = load_frag_b(W, Dm, n0 + 16 * j, kn);
    }
#pragma unroll
    for (int j = 0; j < 4; ++j) acc[j] = wmma_bf16(a_cur, b_cur[j], acc[j]);
    a_cur = a_nxt;
#pragma unroll
    for (int j = 0; j < 4; ++j) b_cur[j] = b_nxt[j];
  }

  const int lane = threadIdx.x & 31;
  const int half = lane >> 4;
  const int c16 = lane & 15;
#pragma unroll
  for (int j = 0; j < 4; ++j) {
#pragma unroll
    for (int r = 0; r < 8; ++r) {
      const int row = m0 + half * 8 + r;          // token index (b*Sq+s)
      const int col = n0 + j * 16 + c16;          // feature index
      const float v = acc[j][r];
      if (MODE == 3) {
        outf[(size_t)row * Dm + col] = v;
      } else {
        const int b = row >> 11;                  // /Sq
        const int s = row & (Sq - 1);
        const int h = col >> 6;                   // /Dh
        const int d = col & (Dh - 1);
        if (MODE == 0)
          outb[((size_t)(b * Hn + h) * Sq + s) * Dh + d] =
              f32_to_bf16(v * 0.125f);            // 1/sqrt(64)
        else if (MODE == 1)
          outb[((size_t)(b * Hn + h) * Sq + s) * Dh + d] = f32_to_bf16(v);
        else  // MODE == 2 : V transposed
          outb[((size_t)(b * Hn + h) * Dh + d) * Sq + s] = f32_to_bf16(v);
      }
    }
  }
}

// ---- Flash attention ------------------------------------------------------
// One block = 4 waves = 4 query tiles of the SAME (b,h): the K/V stream is
// shared, so it is staged once per block into double-buffered LDS tiles via
// GLOBAL_LOAD_ASYNC_TO_LDS_B128 (ASYNCcnt + barrier handshake), then
// consumed as WMMA B-fragments via ds_load.
__global__ __launch_bounds__(128)
void attn_kernel(const unsigned short* __restrict__ Q,   // [B,H,S,Dh] bf16 (pre-scaled)
                 const unsigned short* __restrict__ Kc,  // [B,H,S,Dh] bf16
                 const unsigned short* __restrict__ Vt,  // [B,H,Dh,S] bf16
                 unsigned short* __restrict__ O) {       // [B,S,Dm]   bf16
  const int tid  = threadIdx.x;
  const int wave = tid >> 5;
  const int lane = tid & 31;
  const int half = lane >> 4;
  const int c16  = lane & 15;
  const int qt = blockIdx.x * 4 + wave;   // query tile (16 rows)
  const int bh = blockIdx.y;
  const int b = bh >> 4, h = bh & (Hn - 1);

  const unsigned short* Qbh = Q  + (size_t)bh * Sq * Dh;
  const unsigned short* Kbh = Kc + (size_t)bh * Sq * Dh;
  const unsigned short* Vbh = Vt + (size_t)bh * Dh * Sq;

  __shared__ unsigned short Kst[2][32 * 64];  // keys t0..t0+31, row-major [t][d]
  __shared__ unsigned short Vst[2][64 * 32];  // Vt slice, row-major [d][t-t0]
  __shared__ unsigned short Pst[4][16 * 32];  // per-wave P staging
  unsigned short* Plds = Pst[wave];

  // cooperative stage of one 32-key K/V tile into LDS buffer `buf`
  auto stage = [&](int buf, int t0) {
    // K tile: rows are contiguous -> one flat 4KB block, 256 x 16B chunks
    const unsigned short* ksrc = Kbh + (size_t)t0 * Dh;
#pragma unroll
    for (int c = tid; c < 256; c += 128)
      copy16_to_lds(ksrc + c * 8, Kst[buf] + c * 8);
    // V tile: 64 rows (d) x 32 cols (t), row stride Sq -> 4 chunks per row
#pragma unroll
    for (int c = tid; c < 256; c += 128) {
      const int row = c >> 2;
      const int off = (c & 3) * 8;
      copy16_to_lds(Vbh + (size_t)row * Sq + t0 + off,
                    Vst[buf] + row * 32 + off);
    }
  };

  // Q tile kept resident: 16x64 = two A-frags
  const v16bf qa0 = load_frag_a(Qbh, Dh, qt * 16, 0);
  const v16bf qa1 = load_frag_a(Qbh, Dh, qt * 16, 32);

  v8f zero = {};
  v8f o[4];
#pragma unroll
  for (int j = 0; j < 4; ++j) o[j] = zero;
  float mrow[8], lrow[8];
#pragma unroll
  for (int r = 0; r < 8; ++r) { mrow[r] = -1e30f; lrow[r] = 0.0f; }

  stage(0, 0);  // prologue

  for (int t0 = 0; t0 < Sq; t0 += 32) {
    const int buf = (t0 >> 5) & 1;
    wait_stage_done();      // own async copies complete
    __syncthreads();        // tile `buf` visible to all waves
    if (t0 + 32 < Sq) stage(buf ^ 1, t0 + 32);  // overlap next-tile staging

    const unsigned short* Ks = Kst[buf];
    const unsigned short* Vs = Vst[buf];

    // scores: 16 queries x 32 keys (two 16x16 C tiles).
    // All four B-frags issued before the WMMAs so the ds_loads overlap.
    const v16bf kb00 = load_frag_b(Ks, Dh, 0, 0);
    const v16bf kb01 = load_frag_b(Ks, Dh, 0, 32);
    const v16bf kb10 = load_frag_b(Ks, Dh, 16, 0);
    const v16bf kb11 = load_frag_b(Ks, Dh, 16, 32);
    v8f s0 = zero, s1 = zero;
    s0 = wmma_bf16(qa0, kb00, s0);
    s0 = wmma_bf16(qa1, kb01, s0);
    s1 = wmma_bf16(qa0, kb10, s1);
    s1 = wmma_bf16(qa1, kb11, s1);

    // online softmax: row = half*8 + r, spread across the 16 lanes of a half
    float rmax[8];
#pragma unroll
    for (int r = 0; r < 8; ++r) rmax[r] = fmaxf(s0[r], s1[r]);
#pragma unroll
    for (int off = 1; off < 16; off <<= 1)
#pragma unroll
      for (int r = 0; r < 8; ++r)
        rmax[r] = fmaxf(rmax[r], __shfl_xor(rmax[r], off, 32));

    float scale[8], p0[8], p1[8], rsum[8];
#pragma unroll
    for (int r = 0; r < 8; ++r) {
      const float mnew = fmaxf(mrow[r], rmax[r]);
      scale[r] = __expf(mrow[r] - mnew);
      p0[r] = __expf(s0[r] - mnew);
      p1[r] = __expf(s1[r] - mnew);
      mrow[r] = mnew;
      rsum[r] = p0[r] + p1[r];
    }
#pragma unroll
    for (int off = 1; off < 16; off <<= 1)
#pragma unroll
      for (int r = 0; r < 8; ++r) rsum[r] += __shfl_xor(rsum[r], off, 32);
#pragma unroll
    for (int r = 0; r < 8; ++r) lrow[r] = lrow[r] * scale[r] + rsum[r];
#pragma unroll
    for (int j = 0; j < 4; ++j)
#pragma unroll
      for (int r = 0; r < 8; ++r) o[j][r] *= scale[r];

    // P: C-layout -> LDS (16x32 row-major bf16) -> A-frag
#pragma unroll
    for (int r = 0; r < 8; ++r) {
      const int row = half * 8 + r;
      Plds[row * 32 + c16]      = f32_to_bf16(p0[r]);
      Plds[row * 32 + 16 + c16] = f32_to_bf16(p1[r]);
    }
    asm volatile("s_wait_dscnt 0" ::: "memory");
    const v16bf pa = load_frag_a(Plds, 32, 0, 0);

    // O += P @ V   (B[k=t][n=d] = Vs[d][t], contiguous LDS rows).
    // Batch the V B-frag loads ahead of the WMMAs as well.
    v16bf vb[4];
#pragma unroll
    for (int j = 0; j < 4; ++j) vb[j] = load_frag_b(Vs, 32, j * 16, 0);
#pragma unroll
    for (int j = 0; j < 4; ++j) o[j] = wmma_bf16(pa, vb[j], o[j]);
  }

  // epilogue: normalize and scatter into [B,S,Dm]
#pragma unroll
  for (int r = 0; r < 8; ++r) {
    const float inv = 1.0f / lrow[r];
    const int s = qt * 16 + half * 8 + r;
#pragma unroll
    for (int j = 0; j < 4; ++j) {
      const int col = h * Dh + j * 16 + c16;
      O[((size_t)(b * Sq + s)) * Dm + col] = f32_to_bf16(o[j][r] * inv);
    }
  }
}

// ---------------------------------------------------------------------------

extern "C" void kernel_launch(void* const* d_in, const int* in_sizes, int n_in,
                              void* d_out, int out_size, void* d_ws, size_t ws_size,
                              hipStream_t stream) {
  const float* X  = (const float*)d_in[0];
  const float* Wq = (const float*)d_in[1];
  const float* Wk = (const float*)d_in[2];
  const float* Wv = (const float*)d_in[3];
  const float* Wo = (const float*)d_in[4];

  // workspace layout (bf16 as ushort); total ~50.3 MB
  unsigned short* ws = (unsigned short*)d_ws;
  size_t off = 0;
  unsigned short* Xbf  = ws + off; off += (size_t)BS * Dm;
  unsigned short* Wqbf = ws + off; off += (size_t)Dm * Dm;
  unsigned short* Wkbf = ws + off; off += (size_t)Dm * Dm;
  unsigned short* Wvbf = ws + off; off += (size_t)Dm * Dm;
  unsigned short* Wobf = ws + off; off += (size_t)Dm * Dm;
  unsigned short* Qbf  = ws + off; off += (size_t)BS * Dm;
  unsigned short* Kbf  = ws + off; off += (size_t)BS * Dm;
  unsigned short* Vtbf = ws + off; off += (size_t)BS * Dm;
  unsigned short* Obf  = ws + off; off += (size_t)BS * Dm;
  (void)ws_size; (void)in_sizes; (void)n_in; (void)out_size;

  // 1) convert everything to bf16
  {
    const int nx4 = BS * Dm / 4;
    cvt_f32_bf16_kernel<<<(nx4 + 255) / 256, 256, 0, stream>>>(
        (const float4*)X, (unsigned long long*)Xbf, nx4);
    const int nw4 = Dm * Dm / 4;
    cvt_f32_bf16_kernel<<<(nw4 + 255) / 256, 256, 0, stream>>>(
        (const float4*)Wq, (unsigned long long*)Wqbf, nw4);
    cvt_f32_bf16_kernel<<<(nw4 + 255) / 256, 256, 0, stream>>>(
        (const float4*)Wk, (unsigned long long*)Wkbf, nw4);
    cvt_f32_bf16_kernel<<<(nw4 + 255) / 256, 256, 0, stream>>>(
        (const float4*)Wv, (unsigned long long*)Wvbf, nw4);
    cvt_f32_bf16_kernel<<<(nw4 + 255) / 256, 256, 0, stream>>>(
        (const float4*)Wo, (unsigned long long*)Wobf, nw4);
  }

  // 2) Q/K/V projections (WMMA GEMMs)
  const dim3 gg(BS / 16, Dm / 256);
  gemm_bf16_kernel<0><<<gg, 128, 0, stream>>>(Xbf, Wqbf, Qbf, nullptr);
  gemm_bf16_kernel<1><<<gg, 128, 0, stream>>>(Xbf, Wkbf, Kbf, nullptr);
  gemm_bf16_kernel<2><<<gg, 128, 0, stream>>>(Xbf, Wvbf, Vtbf, nullptr);

  // 3) flash attention: 128 query tiles / 4 per block, B*H batch-heads
  attn_kernel<<<dim3(Sq / 64, Bsz * Hn), 128, 0, stream>>>(Qbf, Kbf, Vtbf, Obf);

  // 4) output projection -> f32 d_out
  gemm_bf16_kernel<3><<<gg, 128, 0, stream>>>(Obf, Wobf, nullptr, (float*)d_out);
}
